// HybridMemory_91233695301908
// MI455X (gfx1250) — matrix operand order repaired and checked by reading, not compiled, not run on hardware
//
#include <hip/hip_runtime.h>
#include <hip/hip_bf16.h>
#include <math.h>

typedef __attribute__((ext_vector_type(16))) _Float16 v16h;
typedef __attribute__((ext_vector_type(8)))  _Float16 v8h;
typedef __attribute__((ext_vector_type(8)))  float    v8f;
typedef __attribute__((ext_vector_type(4)))  float    v4f;

#define M_TOTAL 256
#define N_TOTAL 32768
#define K_TOTAL 2048
#define NTILE   128          // columns per workgroup (8 waves x 16 cols)
#define KSTAGE  32           // K elements per LDS stage (one WMMA K-chunk)
#define NSTAGES (K_TOTAL / KSTAGE)   // 64
#define LDS_ROW 40           // 32 halfs + 8 pad = 80 B stride (20 dwords: bank-spread)
#define BUFHALFS (M_TOTAL * LDS_ROW) // 10240 halfs = 20480 B per buffer
#define TEMP_INV 20.0f       // 1 / 0.05
#define EPS_SOFTMAX 1e-6f
#define EPS_LOG     1e-6f

// ---------------- prep: fp32 -> fp16 A, gather targets ----------------
__global__ __launch_bounds__(256) void hm_prep(const float* __restrict__ inputs,
                                               const int* __restrict__ indexes,
                                               const int* __restrict__ labels,
                                               _Float16* __restrict__ Ah,
                                               int* __restrict__ targets) {
    int i = blockIdx.x * blockDim.x + threadIdx.x;
    if (i < M_TOTAL * K_TOTAL) Ah[i] = (_Float16)inputs[i];
    if (i < M_TOTAL) targets[i] = labels[indexes[i]];
}

// A-fragment gather from LDS (16-bit A 16x32 layout, ISA 7.12.2)
__device__ __forceinline__ v16h load_afrag(const _Float16* Asb, int m, int lr, int h) {
    const _Float16* arow = Asb + (m * 16 + lr) * LDS_ROW + h * 8;
    v8h a0 = *(const v8h*)arow;        // K off+0..7
    v8h a1 = *(const v8h*)(arow + 16); // K off+16..23
    v16h af;
#pragma unroll
    for (int q = 0; q < 8; ++q) { af[q] = a0[q]; af[8 + q] = a1[q]; }
    return af;
}

// ---------------- main GEMM + exp-reduce kernel ----------------
__global__ __launch_bounds__(256) void hm_gemm(const _Float16* __restrict__ Ah,
                                               const float* __restrict__ cf,
                                               const int* __restrict__ targets,
                                               float* __restrict__ partial,
                                               float* __restrict__ tlogit) {
    __shared__ _Float16 As[2 * BUFHALFS]; // 40,960 B double-buffered A stage

    const int tid  = threadIdx.x;
    const int wave = tid >> 5;
    const int lane = tid & 31;
    const int lr   = lane & 15;   // lane within half-wave
    const int h    = lane >> 4;   // half-wave select
    const int col_base = blockIdx.x * NTILE + wave * 16;

    v8f acc[16];
#pragma unroll
    for (int m = 0; m < 16; ++m)
#pragma unroll
        for (int j = 0; j < 8; ++j) acc[m][j] = 0.0f;

    // async staging: 4 x 16B segments per 64B row, 256 threads -> 64 rows/iter
    const int seg  = tid & 3;
    const int row0 = tid >> 2;

    auto issue_async = [&](int s, int buf) {
#pragma unroll
        for (int it = 0; it < 4; ++it) {
            int row = row0 + it * 64;
            unsigned long long g = (unsigned long long)(uintptr_t)
                (Ah + (size_t)row * K_TOTAL + s * KSTAGE + seg * 8);
            unsigned l = (unsigned)(uintptr_t)
                (As + buf * BUFHALFS + row * LDS_ROW + seg * 8);
            asm volatile("global_load_async_to_lds_b128 %0, %1, off"
                         :: "v"(l), "v"(g) : "memory");
        }
    };
    auto issue_b = [&](int s, v4f* braw) {
        const float* bsrc = cf + (size_t)(col_base + lr) * K_TOTAL
                               + s * KSTAGE + h * 8;
#pragma unroll
        for (int q = 0; q < 4; ++q) braw[q] = ((const v4f*)bsrc)[(q & 1) + (q >> 1) * 4];
    };

    // prologue: fill stage 0 + prefetch B for stage 0
    issue_async(0, 0);
    v4f bnext[4];
    issue_b(0, bnext);

    for (int s = 0; s < NSTAGES; ++s) {
        asm volatile("s_wait_asynccnt 0" ::: "memory"); // my LDS fill landed
        __syncthreads();                                // everyone's landed / prev reads done

        // convert this stage's B to f16 (waits loadcnt as needed)
        v16h bfrag;
#pragma unroll
        for (int q = 0; q < 4; ++q) {
            bfrag[q * 4 + 0] = (_Float16)bnext[q][0];
            bfrag[q * 4 + 1] = (_Float16)bnext[q][1];
            bfrag[q * 4 + 2] = (_Float16)bnext[q][2];
            bfrag[q * 4 + 3] = (_Float16)bnext[q][3];
        }
        // kick off next stage's A (async DMA) and B (to VGPRs) behind compute
        if (s + 1 < NSTAGES) {
            issue_async(s + 1, (s + 1) & 1);
            issue_b(s + 1, bnext);
        }

        // 16 M-tiles, software-pipelined A fragments from LDS
        const _Float16* Asb = As + (s & 1) * BUFHALFS;
        v16h afcur = load_afrag(Asb, 0, lr, h);
#pragma unroll
        for (int m = 0; m < 16; ++m) {
            v16h afnext;
            if (m < 15) afnext = load_afrag(Asb, m + 1, lr, h);
            acc[m] = __builtin_amdgcn_wmma_f32_16x16x32_f16(
                false, afcur, false, bfrag, (short)0, acc[m], false, false);
            afcur = afnext;
        }
    }

    // ---- epilogue: exp + 16-lane reduce; per-wave row sums via LDS ----
    __syncthreads();
    float* red = (float*)As; // 8 waves x 256 rows = 8 KB, fits in buffer 0
    const int col = col_base + lr;
#pragma unroll
    for (int m = 0; m < 16; ++m) {
#pragma unroll
        for (int j = 0; j < 8; ++j) {
            int row = m * 16 + h * 8 + j;
            float lg = acc[m][j] * TEMP_INV;
            if (targets[row] == col) tlogit[row] = lg; // unique writer grid-wide
            float e = expf(lg);
            e += __shfl_xor(e, 1, 32);
            e += __shfl_xor(e, 2, 32);
            e += __shfl_xor(e, 4, 32);
            e += __shfl_xor(e, 8, 32);
            if (lr == 0) red[wave * M_TOTAL + row] = e;
        }
    }
    __syncthreads();
    float sum = 0.0f;
#pragma unroll
    for (int w = 0; w < 8; ++w) sum += red[w * M_TOTAL + tid];
    partial[blockIdx.x * M_TOTAL + tid] = sum;   // deterministic, no atomics
}

// ---------------- final: per-row log-softmax at target, mean ----------------
__global__ __launch_bounds__(256) void hm_final(const float* __restrict__ partial,
                                                const float* __restrict__ tlogit,
                                                float* __restrict__ out) {
    __shared__ float sbuf[256];
    int t = threadIdx.x;
    float denom = 0.0f;
    for (int wg = 0; wg < N_TOTAL / NTILE; ++wg)  // fixed order -> deterministic
        denom += partial[wg * M_TOTAL + t];
    float lp = logf(expf(tlogit[t]) / (denom + EPS_SOFTMAX) + EPS_LOG);
    sbuf[t] = lp;
    __syncthreads();
    for (int off = 128; off > 0; off >>= 1) {
        if (t < off) sbuf[t] += sbuf[t + off];
        __syncthreads();
    }
    if (t == 0) out[0] = -sbuf[0] / (float)M_TOTAL;
}

extern "C" void kernel_launch(void* const* d_in, const int* in_sizes, int n_in,
                              void* d_out, int out_size, void* d_ws, size_t ws_size,
                              hipStream_t stream) {
    const float* inputs  = (const float*)d_in[0];
    const float* cf      = (const float*)d_in[2];   // cluster_features
    const int*   indexes = (const int*)d_in[3];
    const int*   labels  = (const int*)d_in[4];

    char* ws = (char*)d_ws;
    _Float16* Ah     = (_Float16*)ws;                                  // 1 MB
    float* partial   = (float*)(ws + 1048576);                         // 256 KB
    float* tlogit    = (float*)(ws + 1048576 + 262144);                // 1 KB
    int*   targets   = (int*)  (ws + 1048576 + 262144 + 1024);         // 1 KB

    hm_prep<<<(M_TOTAL * K_TOTAL) / 256, 256, 0, stream>>>(inputs, indexes, labels, Ah, targets);
    hm_gemm<<<N_TOTAL / NTILE, 256, 0, stream>>>(Ah, cf, targets, partial, tlogit);
    hm_final<<<1, 256, 0, stream>>>(partial, tlogit, (float*)d_out);
}
